// AttentiveReadIn_53480932770345
// MI455X (gfx1250) — matrix operand order repaired and checked by reading, not compiled, not run on hardware
//
#include <hip/hip_runtime.h>
#include <math.h>

#define B_ 2
#define U_ 32
#define V_ 2048
#define IN_ 256
#define STATE_ 512
#define CODE_ 256
#define H_ 8
#define HD_ 64
#define INNER_ 512
#define HID_ 512
#define UH_ 256   // U_*H_

typedef __attribute__((ext_vector_type(16))) __bf16 v16bf;
typedef __attribute__((ext_vector_type(8)))  float  v8f;

__device__ __forceinline__ unsigned short f2bf(float f) {
  unsigned u = __float_as_uint(f);
  u += 0x7FFFu + ((u >> 16) & 1u);
  return (unsigned short)(u >> 16);
}

// --- WMMA fragment gathers from LDS (bf16, 16x16x32) -------------------------
// A 16x32 (MxK): lanes 0-15 m=lane, VGPR0..3 hold K={0,1},{2,3},{4,5},{6,7};
// VGPR4..7 hold K=16..23; lanes 16-31 get +8 on the K base.
__device__ __forceinline__ v16bf lds_fragA(const unsigned short* tile, int ldk) {
  int lane = threadIdx.x & 31;
  int g = lane >> 4;
  const unsigned short* p = tile + (lane & 15) * ldk;
  union { unsigned u[8]; v16bf v; } f;
#pragma unroll
  for (int pi = 0; pi < 8; ++pi) {
    int k = ((pi < 4) ? 0 : 16) + (pi & 3) * 2 + g * 8;
    f.u[pi] = *(const unsigned*)(p + k);
  }
  return f.v;
}
// B 32x16 (KxN): n = lane&15; same K layout as A: VGPR0..3 K=0..7, VGPR4..7 K=16..23 (+8 for lanes 16-31).
// Tile stored transposed in LDS as [n][k], ldk in halfs.
__device__ __forceinline__ v16bf lds_fragB(const unsigned short* tile, int ldk) {
  int lane = threadIdx.x & 31;
  int g = lane >> 4;
  const unsigned short* p = tile + (lane & 15) * ldk;
  union { unsigned u[8]; v16bf v; } f;
#pragma unroll
  for (int pi = 0; pi < 8; ++pi) {
    int k = ((pi < 4) ? 0 : 16) + (pi & 3) * 2 + g * 8;
    f.u[pi] = *(const unsigned*)(p + k);
  }
  return f.v;
}

__device__ __forceinline__ float block_sum256(float v, float* scratch) {
#pragma unroll
  for (int o = 16; o; o >>= 1) v += __shfl_xor(v, o, 32);
  __syncthreads();
  if ((threadIdx.x & 31) == 0) scratch[threadIdx.x >> 5] = v;
  __syncthreads();
  float s = 0.f;
#pragma unroll
  for (int i = 0; i < 8; ++i) s += scratch[i];
  return s;
}

// --- 1: layernorm of sender states -> bf16 ----------------------------------
__global__ void k_ln_senders(const float* __restrict__ s, const float* __restrict__ g,
                             const float* __restrict__ b, unsigned short* __restrict__ out) {
  int row = blockIdx.x * 8 + (threadIdx.x >> 5);
  int lane = threadIdx.x & 31;
  const float* src = s + (size_t)row * IN_;
  float v[8];
  float sum = 0.f, sq = 0.f;
#pragma unroll
  for (int j = 0; j < 8; ++j) {
    v[j] = src[lane + 32 * j];
    sum += v[j]; sq += v[j] * v[j];
  }
#pragma unroll
  for (int o = 16; o; o >>= 1) { sum += __shfl_xor(sum, o, 32); sq += __shfl_xor(sq, o, 32); }
  float mu = sum * (1.f / IN_);
  float var = sq * (1.f / IN_) - mu * mu;
  float rs = rsqrtf(var + 1e-5f);
#pragma unroll
  for (int j = 0; j < 8; ++j) {
    int i = lane + 32 * j;
    out[(size_t)row * IN_ + i] = f2bf((v[j] - mu) * rs * g[i] + b[i]);
  }
}

// --- 2: per-(b,u) prep: r_ln, modulations, q, folded qw ----------------------
__global__ void k_prep(const float* __restrict__ rstates, const float* __restrict__ codes,
                       const float* __restrict__ gr, const float* __restrict__ br,
                       const float* __restrict__ Wq, const float* __restrict__ bq,
                       const float* __restrict__ Cq, const float* __restrict__ Wk,
                       const float* __restrict__ Ck, const float* __restrict__ Cv,
                       const float* __restrict__ Ce, const float* __restrict__ C1,
                       const float* __restrict__ C2,
                       unsigned short* __restrict__ qw,
                       float* __restrict__ sv, float* __restrict__ se,
                       float* __restrict__ s1, float* __restrict__ s2) {
  __shared__ float cds[CODE_];
  __shared__ float rmod[STATE_];
  __shared__ float qv[INNER_];
  __shared__ float sk[IN_];
  __shared__ float scratch[8];
  int bu = blockIdx.x;            // b*U + u
  int tid = threadIdx.x;
  cds[tid] = codes[(size_t)bu * CODE_ + tid];
  const float* xr = rstates + (size_t)bu * STATE_;
  float p0 = xr[tid], p1 = xr[tid + 256];
  float sum = block_sum256(p0 + p1, scratch);
  float sq  = block_sum256(p0 * p0 + p1 * p1, scratch);
  float mu = sum / STATE_;
  float rstd = rsqrtf(sq / STATE_ - mu * mu + 1e-5f);
  for (int s = tid; s < STATE_; s += 256) {
    float r = (xr[s] - mu) * rstd * gr[s] + br[s];
    float d = 0.f;
    const float* c = Cq + (size_t)s * CODE_;
    for (int k = 0; k < CODE_; ++k) d += cds[k] * c[k];
    rmod[s] = r * (1.f + d);
  }
  __syncthreads();
  for (int o = tid; o < INNER_; o += 256) {
    float a = bq[o];
    const float* w = Wq + (size_t)o * STATE_;
    for (int s = 0; s < STATE_; ++s) a += rmod[s] * w[s];
    qv[o] = a;
  }
  {
    int i = tid;                     // IN_ == blockDim
    float d = 0.f;
    const float* c = Ck + (size_t)i * CODE_;
    for (int k = 0; k < CODE_; ++k) d += cds[k] * c[k];
    sk[i] = 1.f + d;
    d = 0.f;
    const float* c2 = Cv + (size_t)i * CODE_;
    for (int k = 0; k < CODE_; ++k) d += cds[k] * c2[k];
    sv[(size_t)bu * IN_ + i] = d;
  }
  for (int o = tid; o < INNER_; o += 256) {
    float d = 0.f;
    const float* c = Ce + (size_t)o * CODE_;
    for (int k = 0; k < CODE_; ++k) d += cds[k] * c[k];
    se[(size_t)bu * INNER_ + o] = d;
    d = 0.f; c = C1 + (size_t)o * CODE_;
    for (int k = 0; k < CODE_; ++k) d += cds[k] * c[k];
    s1[(size_t)bu * STATE_ + o] = d;
    d = 0.f; c = C2 + (size_t)o * CODE_;
    for (int k = 0; k < CODE_; ++k) d += cds[k] * c[k];
    s2[(size_t)bu * HID_ + o] = d;
  }
  __syncthreads();
  // qw[h,i] = 0.125 * scale_k[i] * sum_d q[h*64+d]*Wk[h*64+d, i]
  for (int idx = tid; idx < H_ * IN_; idx += 256) {
    int h = idx >> 8, i = idx & 255;
    float a = 0.f;
    const float* w = Wk + (size_t)(h * HD_) * IN_ + i;
    const float* q = qv + h * HD_;
    for (int d2 = 0; d2 < HD_; ++d2) a += q[d2] * w[(size_t)d2 * IN_];
    qw[(size_t)bu * (H_ * IN_) + idx] = f2bf(a * sk[i] * 0.125f);
  }
}

// --- 3: scores = qw(256x256) @ s_ln^T(256x2048) per batch (bf16 WMMA) --------
__global__ void k_scores(const unsigned short* __restrict__ qw,
                         const unsigned short* __restrict__ sln,
                         float* __restrict__ scores) {
  __shared__ unsigned short At[128 * 32];
  __shared__ unsigned short Bt[128 * 32];
  int bid = blockIdx.x;
  int b  = bid >> 5;
  int mt = (bid >> 4) & 1;
  int nt = bid & 15;
  int tid = threadIdx.x;
  int wave = tid >> 5, lane = tid & 31;
  int wr = wave >> 2, wc = wave & 3;
  const unsigned short* qa = qw  + ((size_t)b * UH_ + mt * 128) * IN_;
  const unsigned short* sb = sln + ((size_t)b * V_  + nt * 128) * IN_;
  v8f z = {0.f,0.f,0.f,0.f,0.f,0.f,0.f,0.f};
  v8f acc[4][2];
#pragma unroll
  for (int i = 0; i < 4; ++i)
#pragma unroll
    for (int j = 0; j < 2; ++j) acc[i][j] = z;
  int srow = tid >> 1, soff = (tid & 1) * 16;
  for (int kk = 0; kk < 8; ++kk) {
    int kb = kk * 32;
    { // stage A (qw rows) and B (s_ln rows, both row-major [row][k])
      const uint4* s4 = (const uint4*)(qa + (size_t)srow * IN_ + kb + soff);
      uint4* d4 = (uint4*)(At + srow * 32 + soff);
      d4[0] = s4[0]; d4[1] = s4[1];
      const uint4* s4b = (const uint4*)(sb + (size_t)srow * IN_ + kb + soff);
      uint4* d4b = (uint4*)(Bt + srow * 32 + soff);
      d4b[0] = s4b[0]; d4b[1] = s4b[1];
    }
    __syncthreads();
    v16bf fa[4], fb[2];
#pragma unroll
    for (int mf = 0; mf < 4; ++mf) fa[mf] = lds_fragA(At + (wr * 64 + mf * 16) * 32, 32);
#pragma unroll
    for (int nf = 0; nf < 2; ++nf) fb[nf] = lds_fragB(Bt + (wc * 32 + nf * 16) * 32, 32);
#pragma unroll
    for (int mf = 0; mf < 4; ++mf)
#pragma unroll
      for (int nf = 0; nf < 2; ++nf)
        acc[mf][nf] = __builtin_amdgcn_wmma_f32_16x16x32_bf16(
            false, fa[mf], false, fb[nf], (short)0, acc[mf][nf], false, false);
    __syncthreads();
  }
  int g = lane >> 4, nlo = lane & 15;
#pragma unroll
  for (int mf = 0; mf < 4; ++mf)
#pragma unroll
    for (int nf = 0; nf < 2; ++nf)
#pragma unroll
      for (int r = 0; r < 8; ++r) {
        int row = mt * 128 + wr * 64 + mf * 16 + r + 8 * g;
        int col = nt * 128 + wc * 32 + nf * 16 + nlo;
        scores[((size_t)b * UH_ + row) * V_ + col] = acc[mf][nf][r];
      }
}

// --- 4: softmax over V per (b,u,h) row -> bf16 attn --------------------------
__global__ void k_softmax(const float* __restrict__ scores, unsigned short* __restrict__ attn) {
  int row = blockIdx.x * 8 + (threadIdx.x >> 5);
  int lane = threadIdx.x & 31;
  const float* sr = scores + (size_t)row * V_;
  float mx = -3.4e38f;
  for (int j = lane; j < V_; j += 32) mx = fmaxf(mx, sr[j]);
#pragma unroll
  for (int o = 16; o; o >>= 1) mx = fmaxf(mx, __shfl_xor(mx, o, 32));
  float sum = 0.f;
  for (int j = lane; j < V_; j += 32) sum += __expf(sr[j] - mx);
#pragma unroll
  for (int o = 16; o; o >>= 1) sum += __shfl_xor(sum, o, 32);
  float inv = 1.f / sum;
  for (int j = lane; j < V_; j += 32)
    attn[(size_t)row * V_ + j] = f2bf(__expf(sr[j] - mx) * inv);
}

// --- 5: ctx = attn(256x2048) @ s_ln(2048x256), K split 4 ways ----------------
__global__ void k_ctx(const unsigned short* __restrict__ attn,
                      const unsigned short* __restrict__ sln,
                      float* __restrict__ ctx) {
  __shared__ unsigned short At[64 * 32];
  __shared__ unsigned short Bt[64 * 32];
  int bid = blockIdx.x;
  int b  = bid >> 6;
  int mt = (bid >> 4) & 3;
  int nt = (bid >> 2) & 3;
  int ks = bid & 3;
  int tid = threadIdx.x;
  int wave = tid >> 5, lane = tid & 31;
  int mf  = wave & 3;
  int nf0 = (wave >> 2) * 2;
  const unsigned short* arow = attn + ((size_t)b * UH_ + mt * 64) * V_;
  v8f z = {0.f,0.f,0.f,0.f,0.f,0.f,0.f,0.f};
  v8f acc[2]; acc[0] = z; acc[1] = z;
  int ar = tid >> 2, aseg = (tid & 3) * 8;
  int bk = tid >> 3, bj = (tid & 7) * 8;
  for (int kk = 0; kk < 16; ++kk) {
    int kb = ks * 512 + kk * 32;
    { // A tile: attn rows, row-major [m][k]
      const uint4* s4 = (const uint4*)(arow + (size_t)ar * V_ + kb + aseg);
      *(uint4*)(At + ar * 32 + aseg) = s4[0];
    }
    { // B tile: s_ln transposed into [n][k]
      union { uint4 q; unsigned short h[8]; } u;
      u.q = *(const uint4*)(sln + ((size_t)b * V_ + kb + bk) * IN_ + nt * 64 + bj);
#pragma unroll
      for (int jj = 0; jj < 8; ++jj) Bt[(bj + jj) * 32 + bk] = u.h[jj];
    }
    __syncthreads();
    v16bf fa = lds_fragA(At + mf * 16 * 32, 32);
#pragma unroll
    for (int j = 0; j < 2; ++j) {
      v16bf fb = lds_fragB(Bt + (nf0 + j) * 16 * 32, 32);
      acc[j] = __builtin_amdgcn_wmma_f32_16x16x32_bf16(
          false, fa, false, fb, (short)0, acc[j], false, false);
    }
    __syncthreads();
  }
  int g = lane >> 4, nlo = lane & 15;
#pragma unroll
  for (int j = 0; j < 2; ++j)
#pragma unroll
    for (int r = 0; r < 8; ++r) {
      int row = mt * 64 + mf * 16 + r + 8 * g;
      int col = nt * 64 + (nf0 + j) * 16 + nlo;
      ctx[(((size_t)b * 4 + ks) * UH_ + row) * IN_ + col] = acc[j][r];
    }
}

// --- 6: per-(b,u) epilogue: msg, exit ModFC, ModFFN --------------------------
__global__ void k_finish(const float* __restrict__ ctx,
                         const float* __restrict__ sv, const float* __restrict__ se,
                         const float* __restrict__ s1, const float* __restrict__ s2,
                         const float* __restrict__ Wv, const float* __restrict__ bv,
                         const float* __restrict__ We, const float* __restrict__ be,
                         const float* __restrict__ ls_attn,
                         const float* __restrict__ gf, const float* __restrict__ bfv,
                         const float* __restrict__ W1, const float* __restrict__ b1,
                         const float* __restrict__ W2, const float* __restrict__ b2,
                         const float* __restrict__ ls_ffn,
                         float* __restrict__ out) {
  __shared__ float ctxh[H_ * IN_];
  __shared__ float msgm[INNER_];
  __shared__ float xv[STATE_];
  __shared__ float rmod[STATE_];
  __shared__ float h1m[HID_];
  __shared__ float scratch[8];
  int bu = blockIdx.x;
  int b = bu >> 5, u = bu & 31;
  int tid = threadIdx.x;
  for (int idx = tid; idx < H_ * IN_; idx += 256) {
    int h = idx >> 8, i = idx & 255;
    float s = 0.f;
#pragma unroll
    for (int ks = 0; ks < 4; ++ks)
      s += ctx[(((size_t)b * 4 + ks) * UH_ + u * H_ + h) * IN_ + i];
    ctxh[idx] = s * (1.f + sv[(size_t)bu * IN_ + i]);
  }
  __syncthreads();
  for (int o = tid; o < INNER_; o += 256) {
    float a = bv[o];
    const float* w = Wv + (size_t)o * IN_;
    const float* c = ctxh + ((o >> 6) << 8);
    for (int i = 0; i < IN_; ++i) a += c[i] * w[i];
    msgm[o] = a * (1.f + se[(size_t)bu * INNER_ + o]);
  }
  __syncthreads();
  for (int s = tid; s < STATE_; s += 256) {
    float a = be[s];
    const float* w = We + (size_t)s * INNER_;
    for (int o = 0; o < INNER_; ++o) a += msgm[o] * w[o];
    xv[s] = a * ls_attn[s];
  }
  __syncthreads();
  float p0 = xv[tid], p1 = xv[tid + 256];
  float sum = block_sum256(p0 + p1, scratch);
  float sq  = block_sum256(p0 * p0 + p1 * p1, scratch);
  float mu = sum / STATE_;
  float rstd = rsqrtf(sq / STATE_ - mu * mu + 1e-5f);
  for (int s = tid; s < STATE_; s += 256)
    rmod[s] = ((xv[s] - mu) * rstd * gf[s] + bfv[s]) * (1.f + s1[(size_t)bu * STATE_ + s]);
  __syncthreads();
  for (int o = tid; o < HID_; o += 256) {
    float a = b1[o];
    const float* w = W1 + (size_t)o * STATE_;
    for (int s = 0; s < STATE_; ++s) a += rmod[s] * w[s];
    float ge = 0.5f * a * (1.f + erff(a * 0.70710678118f));
    h1m[o] = ge * (1.f + s2[(size_t)bu * HID_ + o]);
  }
  __syncthreads();
  for (int s = tid; s < STATE_; s += 256) {
    float a = b2[s];
    const float* w = W2 + (size_t)s * HID_;
    for (int o = 0; o < HID_; ++o) a += h1m[o] * w[o];
    out[(size_t)bu * STATE_ + s] = xv[s] + a * ls_ffn[s];
  }
}

extern "C" void kernel_launch(void* const* d_in, const int* in_sizes, int n_in,
                              void* d_out, int out_size, void* d_ws, size_t ws_size,
                              hipStream_t stream) {
  (void)in_sizes; (void)n_in; (void)out_size; (void)ws_size;
  const float* rstates = (const float*)d_in[0];
  // d_in[1] receiver_signatures: unused by the reference
  const float* codes   = (const float*)d_in[2];
  const float* senders = (const float*)d_in[3];
  const float* ln_s_g  = (const float*)d_in[4];
  const float* ln_s_b  = (const float*)d_in[5];
  const float* ln_r_g  = (const float*)d_in[6];
  const float* ln_r_b  = (const float*)d_in[7];
  const float* Wq = (const float*)d_in[8];
  const float* bq = (const float*)d_in[9];
  const float* Cq = (const float*)d_in[10];
  const float* Wk = (const float*)d_in[11];
  // d_in[12] bk: constant over v -> softmax-invariant, dropped
  const float* Ck = (const float*)d_in[13];
  const float* Wv = (const float*)d_in[14];
  const float* bv = (const float*)d_in[15];
  const float* Cv = (const float*)d_in[16];
  const float* We = (const float*)d_in[17];
  const float* be = (const float*)d_in[18];
  const float* Ce = (const float*)d_in[19];
  const float* ls_attn = (const float*)d_in[20];
  const float* ln_f_g  = (const float*)d_in[21];
  const float* ln_f_b  = (const float*)d_in[22];
  const float* W1 = (const float*)d_in[23];
  const float* b1 = (const float*)d_in[24];
  const float* C1 = (const float*)d_in[25];
  const float* W2 = (const float*)d_in[26];
  const float* b2 = (const float*)d_in[27];
  const float* C2 = (const float*)d_in[28];
  const float* ls_ffn = (const float*)d_in[29];

  char* ws = (char*)d_ws;
  size_t off = 0;
  unsigned short* sln = (unsigned short*)(ws + off); off += (size_t)B_ * V_ * IN_ * 2;
  unsigned short* qw  = (unsigned short*)(ws + off); off += (size_t)B_ * UH_ * IN_ * 2;
  float* scores = (float*)(ws + off);               off += (size_t)B_ * UH_ * V_ * 4;
  unsigned short* attn = (unsigned short*)(ws + off); off += (size_t)B_ * UH_ * V_ * 2;
  float* ctx = (float*)(ws + off);                  off += (size_t)B_ * 4 * UH_ * IN_ * 4;
  float* sv = (float*)(ws + off);                   off += (size_t)B_ * U_ * IN_ * 4;
  float* se = (float*)(ws + off);                   off += (size_t)B_ * U_ * INNER_ * 4;
  float* s1 = (float*)(ws + off);                   off += (size_t)B_ * U_ * STATE_ * 4;
  float* s2 = (float*)(ws + off);                   off += (size_t)B_ * U_ * HID_ * 4;

  dim3 blk(256);
  k_ln_senders<<<dim3((B_ * V_) / 8), blk, 0, stream>>>(senders, ln_s_g, ln_s_b, sln);
  k_prep<<<dim3(B_ * U_), blk, 0, stream>>>(rstates, codes, ln_r_g, ln_r_b,
                                            Wq, bq, Cq, Wk, Ck, Cv, Ce, C1, C2,
                                            qw, sv, se, s1, s2);
  k_scores<<<dim3(B_ * 2 * 16), blk, 0, stream>>>(qw, sln, scores);
  k_softmax<<<dim3((B_ * UH_) / 8), blk, 0, stream>>>(scores, attn);
  k_ctx<<<dim3(B_ * 4 * 4 * 4), blk, 0, stream>>>(attn, sln, ctx);
  k_finish<<<dim3(B_ * U_), blk, 0, stream>>>(ctx, sv, se, s1, s2, Wv, bv, We, be,
                                              ls_attn, ln_f_g, ln_f_b, W1, b1, W2, b2,
                                              ls_ffn, (float*)d_out);
}
